// GeneralGNN_1279900254904
// MI455X (gfx1250) — compile-verified, matrix operand order checked by reading, not run on hardware
//
#include <hip/hip_runtime.h>
#include <hip/hip_bf16.h>
#include <math.h>

typedef __attribute__((ext_vector_type(16))) __bf16 v16bf;
typedef __attribute__((ext_vector_type(8)))  __bf16 v8bf;
typedef __attribute__((ext_vector_type(8)))  float  v8f;

constexpr int Nn = 50000;
constexpr int Ee = 400000;
constexpr int Hh = 128;
constexpr int Bb = 25;
constexpr float kInvSqrtDH = 0.17677669529663687f;  // 1/sqrt(32)

// ---------------- helpers ----------------
__device__ __forceinline__ __bf16 tobf(float x) { return (__bf16)x; }
__device__ __forceinline__ float gelu_f(float x) {
  return 0.5f * x * (1.0f + erff(x * 0.70710678118654752f));
}
__device__ __forceinline__ void wave_lds_wait() {
  asm volatile("s_wait_dscnt 0" ::: "memory");
}
__device__ __forceinline__ void st4(__bf16* p, float4 v) {
  p[0] = tobf(v.x); p[1] = tobf(v.y); p[2] = tobf(v.z); p[3] = tobf(v.w);
}
__device__ __forceinline__ void atomicMaxF(float* a, float v) {
  if (v >= 0.f) atomicMax((int*)a, __float_as_int(v));
  else          atomicMin((unsigned int*)a, __float_as_uint(v));
}
__device__ __forceinline__ v8f bcast8(float b) {
  v8f a;
#pragma unroll
  for (int i = 0; i < 8; ++i) a[i] = b;
  return a;
}
// A/B 16-bit WMMA fragment: per lane, two contiguous 8-elem (16B) chunks.
__device__ __forceinline__ v16bf ldfrag(const __bf16* p0, const __bf16* p1) {
  v8bf a = *(const v8bf*)p0;
  v8bf b = *(const v8bf*)p1;
  v16bf r;
#pragma unroll
  for (int i = 0; i < 8; ++i) { r[i] = a[i]; r[i + 8] = b[i]; }
  return r;
}
// One 16x16 output tile column-block: C += A(16xK, LDS row-major) * Wt_row_n(K)
template <int K>
__device__ __forceinline__ v8f wmma_colblock(const __bf16* As, const __bf16* Wtn,
                                             v8f acc, int half, int m) {
  const __bf16* Arow = As + m * K;
#pragma unroll
  for (int kt = 0; kt < K / 32; ++kt) {
    const int base = kt * 32;
    v16bf a = ldfrag(Arow + base + half * 8, Arow + base + 16 + half * 8);
    v16bf w = ldfrag(Wtn  + base + half * 8, Wtn  + base + 16 + half * 8);
    acc = __builtin_amdgcn_wmma_f32_16x16x32_bf16(false, a, false, w, (short)0,
                                                  acc, false, false);
  }
  return acc;
}

// ---------------- utility kernels ----------------
__global__ void k_fill(float* p, long n, float v) {
  long i = (long)blockIdx.x * blockDim.x + threadIdx.x;
  if (i < n) p[i] = v;
}
// W [din,dout] f32 -> Wt [npad][din] bf16 (rows >= dout zeroed)
__global__ void k_cvt_wt(const float* __restrict__ W, __bf16* __restrict__ Wt,
                         int din, int dout, int npad) {
  int i = blockIdx.x * blockDim.x + threadIdx.x;
  int total = npad * din;
  if (i >= total) return;
  int n = i / din, k = i - n * din;
  float v = (n < dout) ? W[(size_t)k * dout + n] : 0.f;
  Wt[i] = tobf(v);
}
__global__ void k_counts(const int* __restrict__ batch, int* cnt) {
  int i = blockIdx.x * blockDim.x + threadIdx.x;
  if (i < Nn) atomicAdd(&cnt[batch[i]], 1);
}
__global__ void k_finstats(const float* s, const float* q, float* mean,
                           float* rstd, float inv_n) {
  int c = threadIdx.x;
  float m = s[c] * inv_n;
  float v = q[c] * inv_n - m * m;
  mean[c] = m;
  rstd[c] = rsqrtf(fmaxf(v, 0.f) + 1e-5f);
}

// ---------------- kernel 1: attention logits (att_bias MLP) ----------------
__global__ __launch_bounds__(128) void k_att_logits(
    const float* __restrict__ hV, const float* __restrict__ hE,
    const int* __restrict__ src, const int* __restrict__ dst,
    const __bf16* __restrict__ Wt1, const float* __restrict__ b1,
    const __bf16* __restrict__ Wt2, const float* __restrict__ b2,
    const __bf16* __restrict__ Wt3, const float* __restrict__ b3,
    float* __restrict__ logits, float* __restrict__ maxlog) {
  __shared__ __bf16 sA[4][16 * 384];
  __shared__ __bf16 sH1[4][16 * 128];
  __shared__ __bf16 sH2[4][16 * 128];
  __shared__ int sSrc[4][16];
  const int wv = threadIdx.x >> 5, lane = threadIdx.x & 31;
  const int half = lane >> 4, m = lane & 15;
  const long e0 = ((long)blockIdx.x * 4 + wv) * 16;
  if (e0 >= Ee) return;
  __bf16* A = sA[wv];
  if (lane < 16) sSrc[wv][lane] = src[e0 + lane];
  for (int r = 0; r < 16; ++r) {
    long e = e0 + r;
    int s = src[e], d = dst[e];
    float4 v0 = ((const float4*)(hV + (size_t)s * Hh))[lane];
    float4 v1 = ((const float4*)(hE + (size_t)e * Hh))[lane];
    float4 v2 = ((const float4*)(hV + (size_t)d * Hh))[lane];
    st4(A + r * 384 + lane * 4, v0);
    st4(A + r * 384 + 128 + lane * 4, v1);
    st4(A + r * 384 + 256 + lane * 4, v2);
  }
  wave_lds_wait();
  for (int nt = 0; nt < 8; ++nt) {  // 384 -> 128, relu
    int n = nt * 16 + m;
    v8f acc = wmma_colblock<384>(A, Wt1 + n * 384, bcast8(b1[n]), half, m);
#pragma unroll
    for (int i = 0; i < 8; ++i)
      sH1[wv][(half * 8 + i) * 128 + n] = tobf(fmaxf(acc[i], 0.f));
  }
  wave_lds_wait();
  for (int nt = 0; nt < 8; ++nt) {  // 128 -> 128, relu
    int n = nt * 16 + m;
    v8f acc = wmma_colblock<128>(sH1[wv], Wt2 + n * 128, bcast8(b2[n]), half, m);
#pragma unroll
    for (int i = 0; i < 8; ++i)
      sH2[wv][(half * 8 + i) * 128 + n] = tobf(fmaxf(acc[i], 0.f));
  }
  wave_lds_wait();
  {  // 128 -> 4 heads (Wt3 padded to 16 rows)
    v8f acc = wmma_colblock<128>(sH2[wv], Wt3 + m * 128,
                                 bcast8((m < 4) ? b3[m] : 0.f), half, m);
    if (m < 4) {
#pragma unroll
      for (int i = 0; i < 8; ++i) {
        int r = half * 8 + i;
        float lg = acc[i] * kInvSqrtDH;
        logits[(e0 + r) * 4 + m] = lg;
        atomicMaxF(&maxlog[(size_t)sSrc[wv][r] * 4 + m], lg);
      }
    }
  }
}

// ---------------- kernel 2: att_WV MLP + exp + scatter-accumulate ----------------
__global__ __launch_bounds__(128) void k_att_wv(
    const float* __restrict__ hV, const float* __restrict__ hE,
    const int* __restrict__ src, const int* __restrict__ dst,
    const __bf16* __restrict__ Wt1, const float* __restrict__ b1,
    const __bf16* __restrict__ Wt2, const float* __restrict__ b2,
    const __bf16* __restrict__ Wt3, const float* __restrict__ b3,
    const float* __restrict__ logits, const float* __restrict__ maxlog,
    float* __restrict__ sumexp, float* __restrict__ aggV) {
  __shared__ __bf16 sA[4][16 * 256];
  __shared__ __bf16 sH1[4][16 * 128];
  __shared__ __bf16 sH2[4][16 * 128];
  __shared__ float sExp[4][64];
  __shared__ int sSrc[4][16];
  const int wv = threadIdx.x >> 5, lane = threadIdx.x & 31;
  const int half = lane >> 4, m = lane & 15;
  const long e0 = ((long)blockIdx.x * 4 + wv) * 16;
  if (e0 >= Ee) return;
  __bf16* A = sA[wv];
  if (lane < 16) sSrc[wv][lane] = src[e0 + lane];
  for (int r = 0; r < 16; ++r) {  // input = [h_E, h_V[dst]]
    long e = e0 + r;
    int d = dst[e];
    float4 v1 = ((const float4*)(hE + (size_t)e * Hh))[lane];
    float4 v2 = ((const float4*)(hV + (size_t)d * Hh))[lane];
    st4(A + r * 256 + lane * 4, v1);
    st4(A + r * 256 + 128 + lane * 4, v2);
  }
  wave_lds_wait();
  for (int nt = 0; nt < 8; ++nt) {  // 256 -> 128, gelu
    int n = nt * 16 + m;
    v8f acc = wmma_colblock<256>(A, Wt1 + n * 256, bcast8(b1[n]), half, m);
#pragma unroll
    for (int i = 0; i < 8; ++i)
      sH1[wv][(half * 8 + i) * 128 + n] = tobf(gelu_f(acc[i]));
  }
  wave_lds_wait();
  for (int nt = 0; nt < 8; ++nt) {  // 128 -> 128, gelu
    int n = nt * 16 + m;
    v8f acc = wmma_colblock<128>(sH1[wv], Wt2 + n * 128, bcast8(b2[n]), half, m);
#pragma unroll
    for (int i = 0; i < 8; ++i)
      sH2[wv][(half * 8 + i) * 128 + n] = tobf(gelu_f(acc[i]));
  }
  wave_lds_wait();
  {  // exp weights per (row, head); each (r,h) handled exactly once
    int h = lane & 3, rg = lane >> 2;
#pragma unroll
    for (int rr = 0; rr < 2; ++rr) {
      int r = rg * 2 + rr;
      long e = e0 + r;
      int s = sSrc[wv][r];
      float ex = __expf(logits[e * 4 + h] - maxlog[(size_t)s * 4 + h]);
      sExp[wv][r * 4 + h] = ex;
      atomicAdd(&sumexp[(size_t)s * 4 + h], ex);
    }
  }
  wave_lds_wait();
  for (int nt = 0; nt < 8; ++nt) {  // V = 128 -> 128, then scatter exp*V
    int n = nt * 16 + m;
    v8f acc = wmma_colblock<128>(sH2[wv], Wt3 + n * 128, bcast8(b3[n]), half, m);
    int h = nt >> 1;  // head = n >> 5, uniform per nt
#pragma unroll
    for (int i = 0; i < 8; ++i) {
      int r = half * 8 + i;
      float ex = sExp[wv][r * 4 + h];
      atomicAdd(&aggV[(size_t)sSrc[wv][r] * Hh + n], ex * acc[i]);
    }
  }
}

// ---------------- kernel 3: agg/sum @ WO + residual + norm0 stats ----------------
__global__ __launch_bounds__(128) void k_node_wo(
    const float* __restrict__ hV, const float* __restrict__ aggV,
    const float* __restrict__ sumexp, const __bf16* __restrict__ WtO,
    float* __restrict__ x0, float* accS, float* accQ) {
  __shared__ __bf16 sA[4][16 * Hh];
  __shared__ float sbS[Hh], sbQ[Hh];
  const int wv = threadIdx.x >> 5, lane = threadIdx.x & 31;
  const int half = lane >> 4, m = lane & 15;
  sbS[threadIdx.x] = 0.f;
  sbQ[threadIdx.x] = 0.f;
  __syncthreads();
  const int tile = blockIdx.x * 4 + wv;
  const bool active = tile * 16 < Nn;
  if (active) {
    const int t0 = tile * 16;
    __bf16* A = sA[wv];
    for (int r = 0; r < 16; ++r) {
      int v = t0 + r;
      float4 a = make_float4(0.f, 0.f, 0.f, 0.f);
      float inv = 0.f;
      if (v < Nn) {
        a = ((const float4*)(aggV + (size_t)v * Hh))[lane];
        float s = sumexp[(size_t)v * 4 + (lane >> 3)];
        inv = (s > 0.f) ? (1.f / s) : 0.f;
      }
      a.x *= inv; a.y *= inv; a.z *= inv; a.w *= inv;
      st4(A + r * Hh + lane * 4, a);
    }
    wave_lds_wait();
    for (int nt = 0; nt < 8; ++nt) {
      int n = nt * 16 + m;
      v8f acc;
#pragma unroll
      for (int i = 0; i < 8; ++i) {
        int v = t0 + half * 8 + i;
        acc[i] = (v < Nn) ? hV[(size_t)v * Hh + n] : 0.f;  // residual init
      }
      acc = wmma_colblock<Hh>(A, WtO + n * Hh, acc, half, m);
      float ls = 0.f, lq = 0.f;
#pragma unroll
      for (int i = 0; i < 8; ++i) {
        int v = t0 + half * 8 + i;
        if (v < Nn) {
          float x = acc[i];
          x0[(size_t)v * Hh + n] = x;
          ls += x; lq += x * x;
        }
      }
      atomicAdd(&sbS[n], ls);
      atomicAdd(&sbQ[n], lq);
    }
  }
  __syncthreads();
  atomicAdd(&accS[threadIdx.x], sbS[threadIdx.x]);
  atomicAdd(&accQ[threadIdx.x], sbQ[threadIdx.x]);
}

// ---------------- kernel 5: bn0 + dense (128->512->128) + residual + norm1 stats ----
__global__ __launch_bounds__(128) void k_dense(
    const float* __restrict__ x0, const float* __restrict__ mean0,
    const float* __restrict__ rstd0, const float* __restrict__ g0,
    const float* __restrict__ bb0, const __bf16* __restrict__ WtD1,
    const float* __restrict__ bD1, const __bf16* __restrict__ WtD2,
    const float* __restrict__ bD2, float* __restrict__ x1, float* accS,
    float* accQ) {
  __shared__ __bf16 sA[4][16 * 128];
  __shared__ float sV[4][16 * 128];
  __shared__ __bf16 sH[4][16 * 512];
  __shared__ float sbS[Hh], sbQ[Hh];
  const int wv = threadIdx.x >> 5, lane = threadIdx.x & 31;
  const int half = lane >> 4, m = lane & 15;
  sbS[threadIdx.x] = 0.f;
  sbQ[threadIdx.x] = 0.f;
  __syncthreads();
  const int tile = blockIdx.x * 4 + wv;
  const bool active = tile * 16 < Nn;
  if (active) {
    const int t0 = tile * 16;
    __bf16* A = sA[wv];
    const int c = lane * 4;
    for (int r = 0; r < 16; ++r) {
      int v = t0 + r;
      float xs[4] = {0.f, 0.f, 0.f, 0.f};
      if (v < Nn) {
        float4 xv = ((const float4*)(x0 + (size_t)v * Hh))[lane];
        xs[0] = xv.x; xs[1] = xv.y; xs[2] = xv.z; xs[3] = xv.w;
      }
#pragma unroll
      for (int j = 0; j < 4; ++j) {
        float xn = (xs[j] - mean0[c + j]) * rstd0[c + j] * g0[c + j] + bb0[c + j];
        sV[wv][r * 128 + c + j] = xn;       // h_V1 (f32) for residual
        A[r * 128 + c + j] = tobf(xn);
      }
    }
    wave_lds_wait();
    for (int nt = 0; nt < 32; ++nt) {  // 128 -> 512, relu
      int n = nt * 16 + m;
      v8f acc = wmma_colblock<128>(A, WtD1 + n * 128, bcast8(bD1[n]), half, m);
#pragma unroll
      for (int i = 0; i < 8; ++i)
        sH[wv][(half * 8 + i) * 512 + n] = tobf(fmaxf(acc[i], 0.f));
    }
    wave_lds_wait();
    for (int nt = 0; nt < 8; ++nt) {  // 512 -> 128 + residual
      int n = nt * 16 + m;
      v8f acc;
#pragma unroll
      for (int i = 0; i < 8; ++i)
        acc[i] = bD2[n] + sV[wv][(half * 8 + i) * 128 + n];
      acc = wmma_colblock<512>(sH[wv], WtD2 + n * 512, acc, half, m);
      float ls = 0.f, lq = 0.f;
#pragma unroll
      for (int i = 0; i < 8; ++i) {
        int v = t0 + half * 8 + i;
        if (v < Nn) {
          float x = acc[i];
          x1[(size_t)v * Hh + n] = x;
          ls += x; lq += x * x;
        }
      }
      atomicAdd(&sbS[n], ls);
      atomicAdd(&sbQ[n], lq);
    }
  }
  __syncthreads();
  atomicAdd(&accS[threadIdx.x], sbS[threadIdx.x]);
  atomicAdd(&accQ[threadIdx.x], sbQ[threadIdx.x]);
}

// ---------------- bn1 -> h_V2, plus segment sums (batch_id sorted) ----------------
__global__ __launch_bounds__(128) void k_hv2_seg(
    const float* __restrict__ x1, const int* __restrict__ batch,
    const float* mean, const float* rstd, const float* g, const float* b,
    float* __restrict__ hv2, float* __restrict__ segsum) {
  int c = threadIdx.x;
  int r0 = blockIdx.x * 32;
  float mm = mean[c], rs = rstd[c], gg = g[c], bb = b[c];
  float acc = 0.f;
  int cur = -1;
  for (int r = 0; r < 32; ++r) {
    int v = r0 + r;
    if (v >= Nn) break;
    int bt = batch[v];
    if (bt != cur) {
      if (cur >= 0) atomicAdd(&segsum[cur * Hh + c], acc);
      acc = 0.f; cur = bt;
    }
    float y = (x1[(size_t)v * Hh + c] - mm) * rs * gg + bb;
    hv2[(size_t)v * Hh + c] = y;
    acc += y;
  }
  if (cur >= 0) atomicAdd(&segsum[cur * Hh + c], acc);
}

// ---------------- batch gate MLP (25 rows; tiny, plain VALU) ----------------
__global__ __launch_bounds__(128) void k_gate(
    const float* __restrict__ segsum, const int* __restrict__ cnt,
    const float* W1, const float* b1, const float* W2, const float* b2,
    const float* W3, const float* b3, float* __restrict__ gate) {
  __shared__ float s0[Hh], s1[Hh];
  int b = blockIdx.x, t = threadIdx.x;
  s0[t] = segsum[b * Hh + t] / fmaxf((float)cnt[b], 1.f);
  __syncthreads();
  float a = b1[t];
  for (int k = 0; k < Hh; ++k) a += s0[k] * W1[k * Hh + t];
  s1[t] = fmaxf(a, 0.f);
  __syncthreads();
  a = b2[t];
  for (int k = 0; k < Hh; ++k) a += s1[k] * W2[k * Hh + t];
  __syncthreads();
  s0[t] = fmaxf(a, 0.f);
  __syncthreads();
  a = b3[t];
  for (int k = 0; k < Hh; ++k) a += s0[k] * W3[k * Hh + t];
  gate[b * Hh + t] = 1.f / (1.f + __expf(-a));
}

// ---------------- apply node gate -> d_out(hV), accumulate segsum2 ----------------
__global__ __launch_bounds__(128) void k_apply_gateV(
    const float* __restrict__ hv2, const int* __restrict__ batch,
    const float* __restrict__ gateV, float* __restrict__ outV,
    float* __restrict__ segsum2) {
  int c = threadIdx.x;
  int r0 = blockIdx.x * 32;
  float acc = 0.f;
  int cur = -1;
  for (int r = 0; r < 32; ++r) {
    int v = r0 + r;
    if (v >= Nn) break;
    int bt = batch[v];
    if (bt != cur) {
      if (cur >= 0) atomicAdd(&segsum2[cur * Hh + c], acc);
      acc = 0.f; cur = bt;
    }
    float y = hv2[(size_t)v * Hh + c] * gateV[bt * Hh + c];
    outV[(size_t)v * Hh + c] = y;
    acc += y;
  }
  if (cur >= 0) atomicAdd(&segsum2[cur * Hh + c], acc);
}

// ---------------- edge MLP (384->128->128->128 gelu) + residual + stats ----------
__global__ __launch_bounds__(128) void k_edge_mlp(
    const float* __restrict__ hv2, const float* __restrict__ hE,
    const int* __restrict__ src, const int* __restrict__ dst,
    const __bf16* __restrict__ Wt1, const float* __restrict__ b1,
    const __bf16* __restrict__ Wt2, const float* __restrict__ b2,
    const __bf16* __restrict__ Wt3, const float* __restrict__ b3,
    float* __restrict__ outE, float* accS, float* accQ) {
  __shared__ __bf16 sA[4][16 * 384];
  __shared__ __bf16 sH1[4][16 * 128];
  __shared__ __bf16 sH2[4][16 * 128];
  __shared__ float sbS[Hh], sbQ[Hh];
  const int wv = threadIdx.x >> 5, lane = threadIdx.x & 31;
  const int half = lane >> 4, m = lane & 15;
  sbS[threadIdx.x] = 0.f;
  sbQ[threadIdx.x] = 0.f;
  __syncthreads();
  const long e0 = ((long)blockIdx.x * 4 + wv) * 16;
  const bool active = e0 < Ee;
  if (active) {
    __bf16* A = sA[wv];
    for (int r = 0; r < 16; ++r) {  // [h_V2[src], h_E, h_V2[dst]]
      long e = e0 + r;
      int s = src[e], d = dst[e];
      float4 v0 = ((const float4*)(hv2 + (size_t)s * Hh))[lane];
      float4 v1 = ((const float4*)(hE + (size_t)e * Hh))[lane];
      float4 v2 = ((const float4*)(hv2 + (size_t)d * Hh))[lane];
      st4(A + r * 384 + lane * 4, v0);
      st4(A + r * 384 + 128 + lane * 4, v1);
      st4(A + r * 384 + 256 + lane * 4, v2);
    }
    wave_lds_wait();
    for (int nt = 0; nt < 8; ++nt) {
      int n = nt * 16 + m;
      v8f acc = wmma_colblock<384>(A, Wt1 + n * 384, bcast8(b1[n]), half, m);
#pragma unroll
      for (int i = 0; i < 8; ++i)
        sH1[wv][(half * 8 + i) * 128 + n] = tobf(gelu_f(acc[i]));
    }
    wave_lds_wait();
    for (int nt = 0; nt < 8; ++nt) {
      int n = nt * 16 + m;
      v8f acc = wmma_colblock<128>(sH1[wv], Wt2 + n * 128, bcast8(b2[n]), half, m);
#pragma unroll
      for (int i = 0; i < 8; ++i)
        sH2[wv][(half * 8 + i) * 128 + n] = tobf(gelu_f(acc[i]));
    }
    wave_lds_wait();
    for (int nt = 0; nt < 8; ++nt) {  // final + residual h_E
      int n = nt * 16 + m;
      v8f acc;
#pragma unroll
      for (int i = 0; i < 8; ++i)
        acc[i] = b3[n] + hE[(size_t)(e0 + half * 8 + i) * Hh + n];
      acc = wmma_colblock<128>(sH2[wv], Wt3 + n * 128, acc, half, m);
      float ls = 0.f, lq = 0.f;
#pragma unroll
      for (int i = 0; i < 8; ++i) {
        float x = acc[i];
        outE[(size_t)(e0 + half * 8 + i) * Hh + n] = x;
        ls += x; lq += x * x;
      }
      atomicAdd(&sbS[n], ls);
      atomicAdd(&sbQ[n], lq);
    }
  }
  __syncthreads();
  atomicAdd(&accS[threadIdx.x], sbS[threadIdx.x]);
  atomicAdd(&accQ[threadIdx.x], sbQ[threadIdx.x]);
}

// ---------------- final edge bn + gate, in place on d_out ----------------
__global__ void k_edge_final(float* __restrict__ outE, const int* __restrict__ src,
                             const int* __restrict__ batch, const float* meanE,
                             const float* rstdE, const float* gE, const float* bE,
                             const float* __restrict__ gateE) {
  long i = (long)blockIdx.x * blockDim.x + threadIdx.x;
  if (i >= (long)Ee * Hh) return;
  long e = i >> 7;
  int c = (int)(i & 127);
  float y = (outE[i] - meanE[c]) * rstdE[c] * gE[c] + bE[c];
  int bt = batch[src[e]];
  outE[i] = y * gateE[bt * Hh + c];
}

// ================= host =================
extern "C" void kernel_launch(void* const* d_in, const int* in_sizes, int n_in,
                              void* d_out, int out_size, void* d_ws, size_t ws_size,
                              hipStream_t stream) {
  // Flattened input order: jax tree order (dict keys sorted; "W" < "b"; "b" < "g")
  const float* hV = (const float*)d_in[0];
  const float* hE = (const float*)d_in[1];
  const int* eidx = (const int*)d_in[2];
  const int* batch = (const int*)d_in[3];
  const int* src = eidx;
  const int* dst = eidx + Ee;
  const float* W_attWO = (const float*)d_in[4];
  const float* W_wv1 = (const float*)d_in[5];  const float* b_wv1 = (const float*)d_in[6];
  const float* W_wv2 = (const float*)d_in[7];  const float* b_wv2 = (const float*)d_in[8];
  const float* W_wv3 = (const float*)d_in[9];  const float* b_wv3 = (const float*)d_in[10];
  const float* W_ab1 = (const float*)d_in[11]; const float* b_ab1 = (const float*)d_in[12];
  const float* W_ab2 = (const float*)d_in[13]; const float* b_ab2 = (const float*)d_in[14];
  const float* W_ab3 = (const float*)d_in[15]; const float* b_ab3 = (const float*)d_in[16];
  const float* W_eg1 = (const float*)d_in[17]; const float* b_eg1 = (const float*)d_in[18];
  const float* W_eg2 = (const float*)d_in[19]; const float* b_eg2 = (const float*)d_in[20];
  const float* W_eg3 = (const float*)d_in[21]; const float* b_eg3 = (const float*)d_in[22];
  const float* W_vg1 = (const float*)d_in[23]; const float* b_vg1 = (const float*)d_in[24];
  const float* W_vg2 = (const float*)d_in[25]; const float* b_vg2 = (const float*)d_in[26];
  const float* W_vg3 = (const float*)d_in[27]; const float* b_vg3 = (const float*)d_in[28];
  const float* W_d1 = (const float*)d_in[29];  const float* b_d1 = (const float*)d_in[30];
  const float* W_d2 = (const float*)d_in[31];  const float* b_d2 = (const float*)d_in[32];
  const float* W_e1 = (const float*)d_in[33];  const float* b_e1 = (const float*)d_in[34];
  const float* W_e2 = (const float*)d_in[35];  const float* b_e2 = (const float*)d_in[36];
  const float* W_e3 = (const float*)d_in[37];  const float* b_e3 = (const float*)d_in[38];
  const float* n0b = (const float*)d_in[39];   const float* n0g = (const float*)d_in[40];
  const float* n1b = (const float*)d_in[41];   const float* n1g = (const float*)d_in[42];
  const float* neb = (const float*)d_in[43];   const float* neg_ = (const float*)d_in[44];

  float* outV = (float*)d_out;
  float* outE = outV + (size_t)Nn * Hh;

  // workspace layout (floats)
  float* wsf = (float*)d_ws;
  size_t o = 0;
  float* LOG = wsf + o;   o += (size_t)Ee * 4;
  float* MAXL = wsf + o;  o += (size_t)Nn * 4;
  float* SUME = wsf + o;  o += (size_t)Nn * 4;
  float* AGGV = wsf + o;  o += (size_t)Nn * Hh;  // aliased: X1 after k_node_wo
  float* X1 = AGGV;
  float* X0 = wsf + o;    o += (size_t)Nn * Hh;
  float* HV2 = wsf + o;   o += (size_t)Nn * Hh;
  float* SEG1 = wsf + o;  o += Bb * Hh;
  float* SEG2 = wsf + o;  o += Bb * Hh;
  int* CNT = (int*)(wsf + o); o += 32;
  float* GATEV = wsf + o; o += Bb * Hh;
  float* GATEE = wsf + o; o += Bb * Hh;
  float* MEAN0 = wsf + o; o += Hh;
  float* RSTD0 = wsf + o; o += Hh;
  float* MEAN1 = wsf + o; o += Hh;
  float* RSTD1 = wsf + o; o += Hh;
  float* MEANE = wsf + o; o += Hh;
  float* RSTDE = wsf + o; o += Hh;
  float* ACCS = wsf + o;  o += 6 * Hh;  // [s0 q0 s1 q1 se qe]
  float* A0S = ACCS, *A0Q = ACCS + Hh, *A1S = ACCS + 2 * Hh,
       *A1Q = ACCS + 3 * Hh, *AES = ACCS + 4 * Hh, *AEQ = ACCS + 5 * Hh;
  // bf16 transposed weights
  __bf16* wb = (__bf16*)(wsf + o);
  size_t wo = 0;
  __bf16* WT_B1 = wb + wo; wo += 128 * 384;
  __bf16* WT_B2 = wb + wo; wo += 128 * 128;
  __bf16* WT_B3 = wb + wo; wo += 16 * 128;
  __bf16* WT_V1 = wb + wo; wo += 128 * 256;
  __bf16* WT_V2 = wb + wo; wo += 128 * 128;
  __bf16* WT_V3 = wb + wo; wo += 128 * 128;
  __bf16* WT_O  = wb + wo; wo += 128 * 128;
  __bf16* WT_D1 = wb + wo; wo += 512 * 128;
  __bf16* WT_D2 = wb + wo; wo += 128 * 512;
  __bf16* WT_E1 = wb + wo; wo += 128 * 384;
  __bf16* WT_E2 = wb + wo; wo += 128 * 128;
  __bf16* WT_E3 = wb + wo; wo += 128 * 128;

  auto fill = [&](float* p, long n, float v) {
    k_fill<<<(int)((n + 255) / 256), 256, 0, stream>>>(p, n, v);
  };
  auto cvt = [&](const float* W, __bf16* Wt, int din, int dout, int npad) {
    int tot = npad * din;
    k_cvt_wt<<<(tot + 255) / 256, 256, 0, stream>>>(W, Wt, din, dout, npad);
  };

  // ---- init ----
  fill(MAXL, (long)Nn * 4, -3.402823466e38f);
  fill(SUME, (long)Nn * 4, 0.f);
  fill(AGGV, (long)Nn * Hh, 0.f);
  fill(SEG1, Bb * Hh, 0.f);
  fill(SEG2, Bb * Hh, 0.f);
  fill((float*)CNT, 32, 0.f);
  fill(ACCS, 6 * Hh, 0.f);
  // ---- weight conversion ----
  cvt(W_ab1, WT_B1, 384, 128, 128);
  cvt(W_ab2, WT_B2, 128, 128, 128);
  cvt(W_ab3, WT_B3, 128, 4, 16);
  cvt(W_wv1, WT_V1, 256, 128, 128);
  cvt(W_wv2, WT_V2, 128, 128, 128);
  cvt(W_wv3, WT_V3, 128, 128, 128);
  cvt(W_attWO, WT_O, 128, 128, 128);
  cvt(W_d1, WT_D1, 128, 512, 512);
  cvt(W_d2, WT_D2, 512, 128, 128);
  cvt(W_e1, WT_E1, 384, 128, 128);
  cvt(W_e2, WT_E2, 128, 128, 128);
  cvt(W_e3, WT_E3, 128, 128, 128);
  k_counts<<<(Nn + 255) / 256, 256, 0, stream>>>(batch, CNT);

  const int eblocks = Ee / 64;                 // 6250 (exact)
  const int nblocks = (Nn / 16 + 3) / 4;       // 782

  // ---- attention ----
  k_att_logits<<<eblocks, 128, 0, stream>>>(hV, hE, src, dst, WT_B1, b_ab1, WT_B2,
                                            b_ab2, WT_B3, b_ab3, LOG, MAXL);
  k_att_wv<<<eblocks, 128, 0, stream>>>(hV, hE, src, dst, WT_V1, b_wv1, WT_V2,
                                        b_wv2, WT_V3, b_wv3, LOG, MAXL, SUME, AGGV);
  k_node_wo<<<nblocks, 128, 0, stream>>>(hV, AGGV, SUME, WT_O, X0, A0S, A0Q);
  k_finstats<<<1, 128, 0, stream>>>(A0S, A0Q, MEAN0, RSTD0, 1.f / Nn);
  // ---- dense ----
  k_dense<<<nblocks, 128, 0, stream>>>(X0, MEAN0, RSTD0, n0g, n0b, WT_D1, b_d1,
                                       WT_D2, b_d2, X1, A1S, A1Q);
  k_finstats<<<1, 128, 0, stream>>>(A1S, A1Q, MEAN1, RSTD1, 1.f / Nn);
  k_hv2_seg<<<(Nn + 31) / 32, 128, 0, stream>>>(X1, batch, MEAN1, RSTD1, n1g, n1b,
                                                HV2, SEG1);
  // ---- node gating ----
  k_gate<<<Bb, 128, 0, stream>>>(SEG1, CNT, W_vg1, b_vg1, W_vg2, b_vg2, W_vg3,
                                 b_vg3, GATEV);
  k_apply_gateV<<<(Nn + 31) / 32, 128, 0, stream>>>(HV2, batch, GATEV, outV, SEG2);
  k_gate<<<Bb, 128, 0, stream>>>(SEG2, CNT, W_eg1, b_eg1, W_eg2, b_eg2, W_eg3,
                                 b_eg3, GATEE);
  // ---- edge MLP (x_e staged in d_out's h_E region) ----
  k_edge_mlp<<<eblocks, 128, 0, stream>>>(HV2, hE, src, dst, WT_E1, b_e1, WT_E2,
                                          b_e2, WT_E3, b_e3, outE, AES, AEQ);
  k_finstats<<<1, 128, 0, stream>>>(AES, AEQ, MEANE, RSTDE, 1.f / Ee);
  k_edge_final<<<(int)(((long)Ee * Hh + 255) / 256), 256, 0, stream>>>(
      outE, src, batch, MEANE, RSTDE, neg_, neb, GATEE);
}
